// GAT_89704686944355
// MI455X (gfx1250) — matrix operand order repaired and checked by reading, not compiled, not run on hardware
//
#include <hip/hip_runtime.h>
#include <hip/hip_bf16.h>
#include <math.h>

// ---------- types ----------
typedef _Float16 v16h __attribute__((ext_vector_type(16)));
typedef _Float16 v8h  __attribute__((ext_vector_type(8)));
typedef float    v8f  __attribute__((ext_vector_type(8)));

__device__ __forceinline__ float lrelu(float x) { return x > 0.f ? x : 0.2f * x; }

// float atomic max via sign-aware integer atomics (target initialized to -inf)
__device__ __forceinline__ void atomicMaxF(float* addr, float v) {
  if (v >= 0.f) atomicMax((int*)addr, __float_as_int(v));
  else          atomicMin((unsigned int*)addr, __float_as_uint(v));
}

// ---------- small utility kernels ----------
__global__ void k_fill(float* __restrict__ p, float v, long n) {
  long t = (long)blockIdx.x * blockDim.x + threadIdx.x;
  if (t < n) p[t] = v;
}

__global__ void k_cvt_f16(const float* __restrict__ x, _Float16* __restrict__ y, long n) {
  long t = (long)blockIdx.x * blockDim.x + threadIdx.x;
  if (t < n) y[t] = (_Float16)x[t];
}

// W [K][M] f32 -> WT [M][K] f16 (so B fragments load contiguously along K)
__global__ void k_cvt_wT(const float* __restrict__ W, _Float16* __restrict__ WT,
                         int K, int M) {
  int t = blockIdx.x * blockDim.x + threadIdx.x;
  if (t >= K * M) return;
  int k = t / M, m = t - k * M;
  WT[(size_t)m * K + k] = (_Float16)W[t];
}

// ---------- WMMA GEMM: Hf[N,M] = Xh[N,K] @ W[K,M]  (WT is [M][K] f16) ----------
// one wave per 16x16 output tile; K-loop in steps of 32 using v_wmma_f32_16x16x32_f16
__global__ __launch_bounds__(256)
void k_gemm_wmma(const _Float16* __restrict__ Xh, const _Float16* __restrict__ WT,
                 float* __restrict__ Hf, int K, int M, int rowTiles) {
  const int wave = threadIdx.x >> 5;
  const int lane = threadIdx.x & 31;
  const int rowTile = blockIdx.x * (blockDim.x >> 5) + wave;   // wave-uniform
  const int colTile = blockIdx.y;
  if (rowTile >= rowTiles) return;                              // whole-wave exit

  const int hi  = lane >> 4;     // 0: lanes 0-15, 1: lanes 16-31
  const int l15 = lane & 15;
  // A fragment: lane supplies row (rowTile*16 + l15); halves j=0..7 -> K=k0+8*hi+j,
  //             j=8..15 -> K=k0+16+8*hi+(j-8)   (ISA 16-bit A 16x32 layout)
  const _Float16* arow = Xh + (size_t)(rowTile * 16 + l15) * K;
  // B fragment: lane supplies column; halves j=0..7 -> K=k0+8*hi+j, j=8..15 -> K=k0+16+8*hi+(j-8)
  const _Float16* bcol = WT + (size_t)(colTile * 16 + l15) * K;

  v8f c = {};
  for (int k0 = 0; k0 < K; k0 += 32) {
    v8h alo = *(const v8h*)(arow + k0 + hi * 8);
    v8h ahi = *(const v8h*)(arow + k0 + 16 + hi * 8);
    v16h a = __builtin_shufflevector(alo, ahi, 0,1,2,3,4,5,6,7,8,9,10,11,12,13,14,15);
    v8h blo = *(const v8h*)(bcol + k0 + hi * 8);
    v8h bhi = *(const v8h*)(bcol + k0 + 16 + hi * 8);
    v16h b = __builtin_shufflevector(blo, bhi, 0,1,2,3,4,5,6,7,8,9,10,11,12,13,14,15);
    c = __builtin_amdgcn_wmma_f32_16x16x32_f16(false, a, false, b, (short)0, c,
                                               false, false);
  }
  // D layout: VGPR d -> row d + 8*hi, col = l15
  float* out = Hf + (size_t)(rowTile * 16) * M + colTile * 16 + l15;
#pragma unroll
  for (int d = 0; d < 8; ++d)
    out[(size_t)(d + hi * 8) * M] = c[d];
}

// ---------- per-node attention scalars: als/ald [N,8] ----------
__global__ void k_attn(const float* __restrict__ Hf, const float* __restrict__ as,
                       const float* __restrict__ ad, float* __restrict__ als,
                       float* __restrict__ ald, int N, int M, int C) {
  int t = blockIdx.x * blockDim.x + threadIdx.x;
  if (t >= N * 8) return;
  int n = t >> 3, h = t & 7;
  const float* hp = Hf + (size_t)n * M + h * C;
  const float* sp = as + h * C;
  const float* dp = ad + h * C;
  float ss = 0.f, sd = 0.f;
  for (int c = 0; c < C; ++c) { float x = hp[c]; ss += x * sp[c]; sd += x * dp[c]; }
  als[t] = ss;
  ald[t] = sd;
}

// ---------- edge pass 1: segment max over dst ----------
__global__ void k_edge_max(const int* __restrict__ ei, const float* __restrict__ als,
                           const float* __restrict__ ald, float* __restrict__ mx,
                           int E, int Etot) {
  int e = blockIdx.x * blockDim.x + threadIdx.x;
  if (e >= Etot) return;
  int s, d;
  if (e < E) { s = ei[e]; d = ei[E + e]; } else { s = d = e - E; }   // self loops
#pragma unroll
  for (int h = 0; h < 8; ++h) {
    float v = lrelu(als[s * 8 + h] + ald[d * 8 + h]);
    atomicMaxF(&mx[d * 8 + h], v);
  }
}

// ---------- edge pass 2: segment sum of exp(logit - max) ----------
__global__ void k_edge_sum(const int* __restrict__ ei, const float* __restrict__ als,
                           const float* __restrict__ ald, const float* __restrict__ mx,
                           float* __restrict__ den, int E, int Etot) {
  int e = blockIdx.x * blockDim.x + threadIdx.x;
  if (e >= Etot) return;
  int s, d;
  if (e < E) { s = ei[e]; d = ei[E + e]; } else { s = d = e - E; }
#pragma unroll
  for (int h = 0; h < 8; ++h) {
    float v = lrelu(als[s * 8 + h] + ald[d * 8 + h]);
    atomicAdd(&den[d * 8 + h], __expf(v - mx[d * 8 + h]));
  }
}

// ---------- edge pass 3: out[dst] += alpha * h[src]  (thread per edge*head) ----------
__global__ void k_edge_agg(const int* __restrict__ ei, const float* __restrict__ als,
                           const float* __restrict__ ald, const float* __restrict__ mx,
                           const float* __restrict__ den, const float* __restrict__ Hf,
                           float* __restrict__ outb, int E, int Etot, int M, int C) {
  long t = (long)blockIdx.x * blockDim.x + threadIdx.x;
  if (t >= (long)Etot * 8) return;
  int e = (int)(t >> 3), h = (int)(t & 7);
  int s, d;
  if (e < E) { s = ei[e]; d = ei[E + e]; } else { s = d = e - E; }
  float v = lrelu(als[s * 8 + h] + ald[d * 8 + h]);
  float alpha = __expf(v - mx[d * 8 + h]) / (den[d * 8 + h] + 1e-16f);
  const float* hs = Hf + (size_t)s * M + h * C;
  float* od = outb + (size_t)d * M + h * C;
  for (int c = 0; c < C; ++c) atomicAdd(&od[c], alpha * hs[c]);
}

// ---------- layers 1-4: act = lrelu(out + b) ----------
__global__ void k_bias_act(const float* __restrict__ outb, const float* __restrict__ b,
                           float* __restrict__ act, long total, int M) {
  long t = (long)blockIdx.x * blockDim.x + threadIdx.x;
  if (t >= total) return;
  int m = (int)(t % M);
  act[t] = lrelu(outb[t] + b[m]);
}

// ---------- layer 5: mean over heads + bias + log_softmax ----------
__global__ void k_final(const float* __restrict__ outb, const float* __restrict__ b5,
                        float* __restrict__ y, int N) {
  int n = blockIdx.x * blockDim.x + threadIdx.x;
  if (n >= N) return;
  const float* row = outb + (size_t)n * 320;
  float v[40];
  float mxv = -INFINITY;
#pragma unroll 1
  for (int c = 0; c < 40; ++c) {
    float s = 0.f;
#pragma unroll
    for (int h = 0; h < 8; ++h) s += row[h * 40 + c];
    v[c] = s * 0.125f + b5[c];
    mxv = fmaxf(mxv, v[c]);
  }
  float se = 0.f;
#pragma unroll 1
  for (int c = 0; c < 40; ++c) se += __expf(v[c] - mxv);
  float lse = mxv + __logf(se);
#pragma unroll 1
  for (int c = 0; c < 40; ++c) y[n * 40 + c] = v[c] - lse;
}

// ---------- host orchestration ----------
static inline unsigned nblk(long n) { return (unsigned)((n + 255) / 256); }

extern "C" void kernel_launch(void* const* d_in, const int* in_sizes, int n_in,
                              void* d_out, int out_size, void* d_ws, size_t ws_size,
                              hipStream_t stream) {
  const int N    = in_sizes[0] / 128;       // 50000
  const int E    = in_sizes[1] / 2;         // 1600000
  const int Etot = E + N;                   // + self loops
  const int* ei  = (const int*)d_in[1];

  const int Ks[5] = {128, 64, 64, 64, 64};
  const int Ms[5] = {64, 64, 64, 64, 320};
  const int Cs[5] = {8, 8, 8, 8, 40};
  const size_t woff[5] = {0, 8192, 12288, 16384, 20480};   // halves, total 40960

  // workspace layout (all sub-buffers 256B-aligned by construction)
  char* ws = (char*)d_ws;
  _Float16* Wh  = (_Float16*)ws;                              //  81920 B (f16, transposed)
  size_t off = 128 * 1024;
  _Float16* Xh  = (_Float16*)(ws + off); off += (size_t)N * 128 * 2;  // 12.8 MB
  float* Hf     = (float*)(ws + off);    off += (size_t)N * 320 * 4;  // 64 MB
  float* outb   = (float*)(ws + off);    off += (size_t)N * 320 * 4;  // 64 MB
  float* als    = (float*)(ws + off);    off += (size_t)N * 8 * 4;
  float* ald    = (float*)(ws + off);    off += (size_t)N * 8 * 4;
  float* mxb    = (float*)(ws + off);    off += (size_t)N * 8 * 4;
  float* den    = (float*)(ws + off);    off += (size_t)N * 8 * 4;
  float* act    = (float*)(ws + off);    off += (size_t)N * 64 * 4;   // 12.8 MB

  // convert + transpose all weights to f16 once per call
  for (int l = 0; l < 5; ++l) {
    const float* W = (const float*)d_in[2 + 4 * l];
    k_cvt_wT<<<nblk((long)Ks[l] * Ms[l]), 256, 0, stream>>>(W, Wh + woff[l], Ks[l], Ms[l]);
  }

  const float* curx = (const float*)d_in[0];
  for (int l = 0; l < 5; ++l) {
    const int K = Ks[l], M = Ms[l], C = Cs[l];
    const float* a_s = (const float*)d_in[3 + 4 * l];
    const float* a_d = (const float*)d_in[4 + 4 * l];
    const float* bb  = (const float*)d_in[5 + 4 * l];

    // 1) features -> f16
    k_cvt_f16<<<nblk((long)N * K), 256, 0, stream>>>(curx, Xh, (long)N * K);

    // 2) Hf = Xh @ W  via WMMA (wave per 16x16 tile, 8 waves/block)
    const int rowTiles = N / 16;                  // N = 50000 = 3125*16
    dim3 gg((rowTiles + 7) / 8, M / 16);
    k_gemm_wmma<<<gg, 256, 0, stream>>>(Xh, Wh + woff[l], Hf, K, M, rowTiles);

    // 3) per-node attention scalars
    k_attn<<<nblk((long)N * 8), 256, 0, stream>>>(Hf, a_s, a_d, als, ald, N, M, C);

    // 4) init segment buffers
    k_fill<<<nblk((long)N * 8), 256, 0, stream>>>(mxb, -INFINITY, (long)N * 8);
    k_fill<<<nblk((long)N * 8), 256, 0, stream>>>(den, 0.f, (long)N * 8);
    k_fill<<<nblk((long)N * M), 256, 0, stream>>>(outb, 0.f, (long)N * M);

    // 5) three edge sweeps: max, exp-sum, weighted aggregate
    k_edge_max<<<nblk(Etot), 256, 0, stream>>>(ei, als, ald, mxb, E, Etot);
    k_edge_sum<<<nblk(Etot), 256, 0, stream>>>(ei, als, ald, mxb, den, E, Etot);
    k_edge_agg<<<nblk((long)Etot * 8), 256, 0, stream>>>(ei, als, ald, mxb, den,
                                                         Hf, outb, E, Etot, M, C);

    // 6) epilogue
    if (l < 4) {
      k_bias_act<<<nblk((long)N * M), 256, 0, stream>>>(outb, bb, act, (long)N * M, M);
      curx = act;
    } else {
      k_final<<<nblk(N), 256, 0, stream>>>(outb, bb, (float*)d_out, N);
    }
  }
}
